// GraphSAGE_87720412054178
// MI455X (gfx1250) — matrix operand order repaired and checked by reading, not compiled, not run on hardware
//
#include <hip/hip_runtime.h>

typedef __attribute__((ext_vector_type(16))) __bf16 v16bf;
typedef __attribute__((ext_vector_type(8)))  float  v8f;

#define NNODES 10000
#define NEDGES 160000
#define DIM    1024
#define KCAT   2048
#define MPAD   10112   // 79 * 128 tiles of M

union FragB16 { uint4 u[2]; v16bf v; };
union Pack4   { uint2 u; __bf16 h[4]; };

__device__ __forceinline__ void atomAddF(float* p, float v) {
  __hip_atomic_fetch_add(p, v, __ATOMIC_RELAXED, __HIP_MEMORY_SCOPE_AGENT);
}

// ---------------- utility kernels ----------------

__global__ void zero_f32(float* __restrict__ p, int n) {
  for (int i = blockIdx.x * blockDim.x + threadIdx.x; i < n;
       i += gridDim.x * blockDim.x)
    p[i] = 0.0f;
}

__global__ void deg_count(const int* __restrict__ dst, float* __restrict__ deg) {
  int e = blockIdx.x * blockDim.x + threadIdx.x;
  if (e < NEDGES) atomAddF(&deg[dst[e]], 1.0f);
}

// Pack [W_self ; W_neigh] (each [K=1024][N=1024] row-major f32) into a
// transposed bf16 block Wp[N=1024][K=2048] so WMMA B-fragment loads are
// two contiguous 16B chunks per lane.
__global__ void pack_weights(const float* __restrict__ Wself,
                             const float* __restrict__ Wneigh,
                             __bf16* __restrict__ Wp) {
  int id = blockIdx.x * blockDim.x + threadIdx.x;  // exactly DIM*KCAT threads
  int n = id & (DIM - 1);
  int k = id >> 10;
  float v = (k < DIM) ? Wself[(size_t)k * DIM + n]
                      : Wneigh[(size_t)(k - DIM) * DIM + n];
  Wp[(size_t)n * KCAT + k] = (__bf16)v;
}

// x = emb[node[i]] -> bf16 into self half of concatenated activations
__global__ __launch_bounds__(256) void gather_embed(
    const int* __restrict__ node, const float* __restrict__ emb,
    __bf16* __restrict__ Xcat) {
  const int i = blockIdx.x;
  const int t = threadIdx.x;
  const int idx = node[i];
  float4 e4 = ((const float4*)(emb + (size_t)idx * DIM))[t];
  Pack4 p;
  p.h[0] = (__bf16)e4.x; p.h[1] = (__bf16)e4.y;
  p.h[2] = (__bf16)e4.z; p.h[3] = (__bf16)e4.w;
  *(uint2*)(Xcat + (size_t)i * KCAT + t * 4) = p.u;
}

// agg[dst[e]] += Xself[src[e]]  (bf16 gather, f32 atomic scatter-add)
__global__ __launch_bounds__(256) void aggregate_bf16(
    const __bf16* __restrict__ Xself, const int* __restrict__ src,
    const int* __restrict__ dst, float* __restrict__ agg) {
  const int e = blockIdx.x;
  const int s = src[e];
  const int d = dst[e];
  const int t = threadIdx.x;
  Pack4 p;
  p.u = *(const uint2*)(Xself + (size_t)s * KCAT + t * 4);
  float* ap = agg + (size_t)d * DIM + t * 4;
#pragma unroll
  for (int k = 0; k < 4; ++k) atomAddF(ap + k, (float)p.h[k]);
}

// Xcat[:, D:2D] = bf16(agg / max(deg,1))
__global__ __launch_bounds__(256) void normalize_neigh(
    const float* __restrict__ agg, const float* __restrict__ deg,
    __bf16* __restrict__ Xcat) {
  const int i = blockIdx.x;
  const int t = threadIdx.x;
  const float r = 1.0f / fmaxf(deg[i], 1.0f);
  float4 a4 = ((const float4*)(agg + (size_t)i * DIM))[t];
  Pack4 p;
  p.h[0] = (__bf16)(a4.x * r); p.h[1] = (__bf16)(a4.y * r);
  p.h[2] = (__bf16)(a4.z * r); p.h[3] = (__bf16)(a4.w * r);
  *(uint2*)(Xcat + (size_t)i * KCAT + DIM + t * 4) = p.u;
}

// ---------------- WMMA GEMM ----------------
// C[M=10000, N=1024] = A[M][K=2048](bf16) * Wp[N][K](bf16, N-major) + bias
// Block: 256 threads = 8 waves as 2(M) x 4(N); block tile 128x256.
// Wave: 4x4 grid of v_wmma_f32_16x16x32_bf16 -> 64x64 per wave.
template <bool RELU, bool OUTBF16>
__global__ __launch_bounds__(256) void gemm_bf16_wmma(
    const __bf16* __restrict__ A, const __bf16* __restrict__ Wp,
    const float* __restrict__ bias, float* __restrict__ outF,
    __bf16* __restrict__ outB, int mRows, int outStride) {
  const int lane  = threadIdx.x & 31;
  const int wave  = threadIdx.x >> 5;
  const int waveM = wave >> 2;       // 0..1
  const int waveN = wave & 3;        // 0..3
  const int lrow  = lane & 15;
  const int lhalf = lane >> 4;       // 0 or 1
  const int rowBase = blockIdx.x * 128 + waveM * 64;
  const int colBase = blockIdx.y * 256 + waveN * 64;

  v8f acc[4][4];
#pragma unroll
  for (int i = 0; i < 4; ++i)
#pragma unroll
    for (int j = 0; j < 4; ++j) { v8f z = {}; acc[i][j] = z; }

  // A fragment (16x32 bf16): lanes 0-15 hold K [0..8)+[16..24), lanes 16-31
  // hold K [8..16)+[24..32)  -> two 16B chunks at +0 and +16 elements.
  const __bf16* aRow[4];
#pragma unroll
  for (int i = 0; i < 4; ++i)
    aRow[i] = A + (size_t)(rowBase + 16 * i + lrow) * KCAT + lhalf * 8;
  // B fragment (32x16 bf16, stored N-major): same K order as A -> lanes 0-15
  // hold K [0..8)+[16..24), lanes 16-31 hold K [8..16)+[24..32).
  const __bf16* bRow[4];
#pragma unroll
  for (int j = 0; j < 4; ++j)
    bRow[j] = Wp + (size_t)(colBase + 16 * j + lrow) * KCAT + lhalf * 8;

  for (int k0 = 0; k0 < KCAT; k0 += 32) {
    FragB16 a[4], b[4];
#pragma unroll
    for (int i = 0; i < 4; ++i) {
      a[i].u[0] = *(const uint4*)(aRow[i] + k0);
      a[i].u[1] = *(const uint4*)(aRow[i] + k0 + 16);
    }
#pragma unroll
    for (int j = 0; j < 4; ++j) {
      b[j].u[0] = *(const uint4*)(bRow[j] + k0);
      b[j].u[1] = *(const uint4*)(bRow[j] + k0 + 16);
    }
    if (k0 + 64 < KCAT) {
#pragma unroll
      for (int j = 0; j < 4; ++j) __builtin_prefetch(bRow[j] + k0 + 64, 0, 0);
    }
#pragma unroll
    for (int i = 0; i < 4; ++i)
#pragma unroll
      for (int j = 0; j < 4; ++j)
        acc[i][j] = __builtin_amdgcn_wmma_f32_16x16x32_bf16(
            false, a[i].v, false, b[j].v, (short)0, acc[i][j], false, false);
  }

  // D layout: VGPR v -> M = v (+8 for lanes 16-31), N = lane%16
#pragma unroll
  for (int j = 0; j < 4; ++j) {
    const int col = colBase + 16 * j + lrow;
    const float bcol = bias[col];
#pragma unroll
    for (int i = 0; i < 4; ++i) {
#pragma unroll
      for (int v = 0; v < 8; ++v) {
        const int row = rowBase + 16 * i + lhalf * 8 + v;
        if (row < mRows) {
          float val = acc[i][j][v] + bcol;
          if (RELU) val = fmaxf(val, 0.0f);
          if (OUTBF16)
            outB[(size_t)row * outStride + col] = (__bf16)val;
          else
            outF[(size_t)row * outStride + col] = val;
        }
      }
    }
  }
}

// ---------------- launch ----------------

extern "C" void kernel_launch(void* const* d_in, const int* in_sizes, int n_in,
                              void* d_out, int out_size, void* d_ws,
                              size_t ws_size, hipStream_t stream) {
  const int*   node = (const int*)d_in[0];
  const int*   src  = (const int*)d_in[1];
  const int*   dst  = (const int*)d_in[2];
  const float* emb  = (const float*)d_in[3];
  const float* W1s  = (const float*)d_in[4];
  const float* W1n  = (const float*)d_in[5];
  const float* b1   = (const float*)d_in[6];
  const float* W2s  = (const float*)d_in[7];
  const float* W2n  = (const float*)d_in[8];
  const float* b2   = (const float*)d_in[9];
  float* out = (float*)d_out;
  (void)in_sizes; (void)n_in; (void)out_size; (void)ws_size;

  char* ws = (char*)d_ws;
  size_t off = 0;
  auto take = [&](size_t bytes) -> void* {
    void* p = ws + off;
    off += (bytes + 255) & ~(size_t)255;
    return p;
  };
  __bf16* XA  = (__bf16*)take((size_t)MPAD * KCAT * 2);  // layer1 [x | h_neigh1]
  __bf16* XB  = (__bf16*)take((size_t)MPAD * KCAT * 2);  // layer2 [h1 | h_neigh2]
  __bf16* Wp1 = (__bf16*)take((size_t)DIM * KCAT * 2);
  __bf16* Wp2 = (__bf16*)take((size_t)DIM * KCAT * 2);
  float*  agg = (float*)take((size_t)NNODES * DIM * 4);
  float*  deg = (float*)take((size_t)NNODES * 4);

  const dim3 blk(256);
  const dim3 ggrid(MPAD / 128, DIM / 256);  // 79 x 4

  zero_f32<<<512, blk, 0, stream>>>(deg, NNODES);
  zero_f32<<<8192, blk, 0, stream>>>(agg, NNODES * DIM);
  deg_count<<<(NEDGES + 255) / 256, blk, 0, stream>>>(dst, deg);
  pack_weights<<<(DIM * KCAT) / 256, blk, 0, stream>>>(W1s, W1n, Wp1);
  pack_weights<<<(DIM * KCAT) / 256, blk, 0, stream>>>(W2s, W2n, Wp2);

  // layer 1
  gather_embed<<<NNODES, blk, 0, stream>>>(node, emb, XA);
  aggregate_bf16<<<NEDGES, blk, 0, stream>>>(XA, src, dst, agg);
  normalize_neigh<<<NNODES, blk, 0, stream>>>(agg, deg, XA);
  gemm_bf16_wmma<true, true><<<ggrid, blk, 0, stream>>>(
      XA, Wp1, b1, nullptr, XB, NNODES, KCAT);  // relu(h1) -> XB self half

  // layer 2
  zero_f32<<<8192, blk, 0, stream>>>(agg, NNODES * DIM);
  aggregate_bf16<<<NEDGES, blk, 0, stream>>>(XB, src, dst, agg);
  normalize_neigh<<<NNODES, blk, 0, stream>>>(agg, deg, XB);
  gemm_bf16_wmma<false, false><<<ggrid, blk, 0, stream>>>(
      XB, Wp2, b2, out, nullptr, NNODES, DIM);
}
